// TENext_block_46634754900264
// MI455X (gfx1250) — matrix operand order repaired and checked by reading, not compiled, not run on hardware
//
#include <hip/hip_runtime.h>
#include <hip/hip_bf16.h>

typedef __attribute__((ext_vector_type(16))) _Float16 v16h;
typedef __attribute__((ext_vector_type(8)))  _Float16 v8h;
typedef __attribute__((ext_vector_type(8)))  float    v8f;
typedef __attribute__((ext_vector_type(2)))  int      v2i;

#define KOFF 343
#define D2   32
#define PLN  128
#define KC   7            // 343 = 49 * 7 -> uniform chunks
#define NCHUNK 49
#define CHW  (KC * D2 * D2)        // halves per chunk = 7168

#if defined(__has_builtin)
#if __has_builtin(__builtin_amdgcn_global_load_async_to_lds_b64)
#define HAVE_ASYNC 1
#endif
#endif

// ---------------- kernel 0: w_conv fp32 -> f16 ----------------
__global__ __launch_bounds__(256) void wconv_to_f16(const float* __restrict__ w,
                                                    _Float16* __restrict__ w16, int n) {
    int i = blockIdx.x * 256 + threadIdx.x;
    if (i < n) w16[i] = (_Float16)w[i];
}

// ---------------- kernel 1: h16 = LN(feats @ w1), f16 out ----------------
__global__ __launch_bounds__(256) void gemm1_ln(const float* __restrict__ feats,
                                                const float* __restrict__ w1,
                                                const float* __restrict__ g,
                                                const float* __restrict__ b,
                                                _Float16* __restrict__ h16, int n) {
    int t = blockIdx.x * 256 + threadIdx.x;
    int i = t >> 5;
    int j = t & 31;
    if (i >= n) return;
    const float* fr = feats + (size_t)i * PLN;
    float s = 0.f;
#pragma unroll 8
    for (int k = 0; k < PLN; ++k) s = fmaf(fr[k], w1[k * D2 + j], s);
    float sum = s, sq = s * s;
#pragma unroll
    for (int off = 16; off; off >>= 1) {
        sum += __shfl_xor(sum, off, 32);
        sq  += __shfl_xor(sq,  off, 32);
    }
    float m = sum * (1.f / 32.f);
    float v = sq * (1.f / 32.f) - m * m;
    float r = rsqrtf(v + 1e-5f);
    float o = (s - m) * r * g[j] + b[j];
    h16[(size_t)i * D2 + j] = (_Float16)o;
}

// ---- LDS weight staging: async (CDNA5) with sync fallback ----
__device__ __forceinline__ void stage_chunk(const _Float16* __restrict__ wsrc,
                                            _Float16* __restrict__ ldst, int tid) {
#if defined(HAVE_ASYNC)
    const v2i* s = (const v2i*)wsrc;
    v2i* d = (v2i*)ldst;
#pragma unroll
    for (int t = 0; t < 7; ++t) {
        int e = tid + t * 256;
        __builtin_amdgcn_global_load_async_to_lds_b64(
            (__attribute__((address_space(1))) v2i*)(v2i*)(s + e),
            (__attribute__((address_space(3))) v2i*)(d + e),
            0, 0);
    }
#else
    const unsigned long long* s = (const unsigned long long*)wsrc;
    unsigned long long* d = (unsigned long long*)ldst;
#pragma unroll
    for (int t = 0; t < 7; ++t) {
        int e = tid + t * 256;
        d[e] = s[e];
    }
#endif
}

__device__ __forceinline__ void wait_stage() {
#if defined(HAVE_ASYNC)
    asm volatile("s_wait_asynccnt 0x0" ::: "memory");
#endif
    __syncthreads();
}

// ---- A-fragment gather (16x32 f16 WMMA layout) ----
__device__ __forceinline__ v16h load_afrag(const _Float16* __restrict__ hb, int idx) {
    v16h a = {};
    if (idx >= 0) {
        const _Float16* hrow = hb + (size_t)idx * D2;
        union { v16h v; v8h h[2]; } u;
        u.h[0] = *(const v8h*)(hrow);        // K = kb .. kb+7
        u.h[1] = *(const v8h*)(hrow + 16);   // K = kb+16 .. kb+23
        a = u.v;
    }
    return a;
}

// ---------------- kernel 2: sparse conv via WMMA ----------------
// 8 waves/block, 2 tiles (32 points) per wave; weights double-buffered in LDS.
__global__ __launch_bounds__(256) void sparse_conv_wmma(const _Float16* __restrict__ h16,
                                                        const int* __restrict__ nbr,
                                                        const _Float16* __restrict__ w16,
                                                        float* __restrict__ y, int n) {
    __shared__ _Float16 lbuf[2][CHW];   // 2 x 14 KB

    int tid  = threadIdx.x;
    int lane = tid & 31;
    int wid  = tid >> 5;

    int tile0 = (blockIdx.x * 8 + wid) * 2;
    int tile1 = tile0 + 1;

    int m  = lane & 15;
    int kb = (lane < 16) ? 0 : 8;                 // A K-base for this lane
    const _Float16* hb = h16 + kb;

    size_t pt0 = (size_t)tile0 * 16 + m;
    size_t pt1 = (size_t)tile1 * 16 + m;
    bool ok0 = pt0 < (size_t)n;
    bool ok1 = pt1 < (size_t)n;
    const int* np0 = nbr + pt0 * KOFF;
    const int* np1 = nbr + pt1 * KOFF;

    const _Float16* wrowL = (const _Float16*)lbuf + (lane & 15); // column N=lane%16

    v8f acc00 = {}, acc01 = {}, acc10 = {}, acc11 = {};

    // prologue: stage chunk 0 into buffer 0
    stage_chunk(w16, lbuf[0], tid);
    wait_stage();

    for (int c = 0; c < NCHUNK; ++c) {
        int cur = c & 1;
        if (c + 1 < NCHUNK)
            stage_chunk(w16 + (size_t)(c + 1) * CHW, lbuf[(c + 1) & 1], tid);

        const _Float16* wbase = wrowL + (size_t)cur * CHW;
        int k0 = c * KC;
#pragma unroll
        for (int kk = 0; kk < KC; ++kk) {
            int k = k0 + kk;
            int idx0 = ok0 ? np0[k] : -1;
            int idx1 = ok1 ? np1[k] : -1;

            v16h a0 = load_afrag(hb, idx0);
            v16h a1 = load_afrag(hb, idx1);

            const _Float16* wk = wbase + kk * (D2 * D2);
            v16h b0, b1;
#pragma unroll
            for (int s = 0; s < 16; ++s) {
                int kr = kb + (s & 7) + ((s >> 3) << 4);
                b0[s] = wk[kr * D2];        // N = 0..15
                b1[s] = wk[kr * D2 + 16];   // N = 16..31
            }

            acc00 = __builtin_amdgcn_wmma_f32_16x16x32_f16(
                false, a0, false, b0, (short)0, acc00, false, false);
            acc01 = __builtin_amdgcn_wmma_f32_16x16x32_f16(
                false, a0, false, b1, (short)0, acc01, false, false);
            acc10 = __builtin_amdgcn_wmma_f32_16x16x32_f16(
                false, a1, false, b0, (short)0, acc10, false, false);
            acc11 = __builtin_amdgcn_wmma_f32_16x16x32_f16(
                false, a1, false, b1, (short)0, acc11, false, false);
        }
        wait_stage();
    }

    int ncol  = lane & 15;
    int mbase = (lane < 16) ? 0 : 8;
#pragma unroll
    for (int r = 0; r < 8; ++r) {
        size_t row0 = (size_t)tile0 * 16 + mbase + r;
        size_t row1 = (size_t)tile1 * 16 + mbase + r;
        if (row0 < (size_t)n) {
            y[row0 * D2 + ncol]      = acc00[r];
            y[row0 * D2 + 16 + ncol] = acc01[r];
        }
        if (row1 < (size_t)n) {
            y[row1 * D2 + ncol]      = acc10[r];
            y[row1 * D2 + 16 + ncol] = acc11[r];
        }
    }
}

// ---------------- kernel 3: out = gelu(LN(y @ w2)) + feats ----------------
__global__ __launch_bounds__(128) void gemm2_ln_gelu(const float* __restrict__ y,
                                                     const float* __restrict__ w2,
                                                     const float* __restrict__ g,
                                                     const float* __restrict__ b,
                                                     const float* __restrict__ feats,
                                                     float* __restrict__ out, int n) {
    int i = blockIdx.x;
    int j = threadIdx.x;
    if (i >= n) return;
    const float* yr = y + (size_t)i * D2;
    float s = 0.f;
#pragma unroll
    for (int k = 0; k < D2; ++k) s = fmaf(yr[k], w2[k * PLN + j], s);
    float sum = s, sq = s * s;
#pragma unroll
    for (int off = 16; off; off >>= 1) {
        sum += __shfl_xor(sum, off, 32);
        sq  += __shfl_xor(sq,  off, 32);
    }
    __shared__ float s1[4], s2[4];
    int w = j >> 5;
    if ((j & 31) == 0) { s1[w] = sum; s2[w] = sq; }
    __syncthreads();
    sum = s1[0] + s1[1] + s1[2] + s1[3];
    sq  = s2[0] + s2[1] + s2[2] + s2[3];
    float m = sum * (1.f / 128.f);
    float v = sq * (1.f / 128.f) - m * m;
    float r = rsqrtf(v + 1e-5f);
    float x = (s - m) * r * g[j] + b[j];
    float ge = 0.5f * x * (1.f + erff(x * 0.70710678118654752f));
    out[(size_t)i * PLN + j] = ge + feats[(size_t)i * PLN + j];
}

extern "C" void kernel_launch(void* const* d_in, const int* in_sizes, int n_in,
                              void* d_out, int out_size, void* d_ws, size_t ws_size,
                              hipStream_t stream) {
    const float* feats = (const float*)d_in[0];
    const int*   nbr   = (const int*)d_in[1];
    const float* w1    = (const float*)d_in[2];
    const float* ln1g  = (const float*)d_in[3];
    const float* ln1b  = (const float*)d_in[4];
    const float* wconv = (const float*)d_in[5];
    const float* w2    = (const float*)d_in[6];
    const float* ln2g  = (const float*)d_in[7];
    const float* ln2b  = (const float*)d_in[8];
    float* out = (float*)d_out;

    int n  = in_sizes[0] / PLN;                // 100000
    int nw = in_sizes[5];                      // 343*32*32

    char* ws = (char*)d_ws;
    _Float16* w16 = (_Float16*)ws;                          // 702 KB
    _Float16* h16 = (_Float16*)(ws + (size_t)1 * 1048576);  // 6.4 MB
    float*    yy  = (float*)   (ws + (size_t)8 * 1048576);  // 12.8 MB

    wconv_to_f16<<<(nw + 255) / 256, 256, 0, stream>>>(wconv, w16, nw);

    int b1 = (n * D2 + 255) / 256;
    gemm1_ln<<<b1, 256, 0, stream>>>(feats, w1, ln1g, ln1b, h16, n);

    int ntiles = (n + 15) / 16;
    int b2 = (ntiles + 15) / 16;               // 16 tiles per block (2 per wave)
    sparse_conv_wmma<<<b2, 256, 0, stream>>>(h16, nbr, w16, yy, n);

    gemm2_ln_gelu<<<n, 128, 0, stream>>>(yy, w2, ln2g, ln2b, feats, out, n);
}
